// TrajGRU_cell_6141803233715
// MI455X (gfx1250) — compile-verified, matrix operand order checked by reading, not run on hardware
//
#include <hip/hip_runtime.h>
#include <hip/hip_bf16.h>

typedef __attribute__((ext_vector_type(16))) __bf16 v16bf;
typedef __attribute__((ext_vector_type(8)))  float  v8f;
typedef __attribute__((ext_vector_type(4)))  unsigned int u32x4;
typedef __attribute__((ext_vector_type(8)))  int i32x8;
typedef __attribute__((ext_vector_type(4)))  int i32x4;

#ifndef __has_builtin
#define __has_builtin(x) 0
#endif
#if __has_builtin(__builtin_amdgcn_tensor_load_to_lds) && \
    __has_builtin(__builtin_amdgcn_s_wait_tensorcnt)
#define USE_TDM 1
#else
#define USE_TDM 0
#endif

constexpr int Bn = 8, Cin = 64, NFc = 64, Hh = 96, Ww = 96, Lc = 13;
constexpr int HW = Hh * Ww;
constexpr int Npix = Bn * HW;        // 73728 GEMM columns
constexpr int NTILES = Npix / 16;    // 4608 column tiles (exactly 1152 blocks x 4 waves)
constexpr float NEG_SLOPE = 0.2f;

__device__ __forceinline__ int koff_of(int lane, int e) {
  // K index inside a 32-wide chunk for 16-bit A/B fragments (ISA 7.12.2):
  // VGPR0..3 hold K pairs 0..7 (lanes 0-15) / 8..15 (lanes 16-31),
  // VGPR4..7 hold the same pattern +16.
  return ((e >> 3) << 4) + ((lane & 16) >> 1) + (((e >> 1) & 3) << 1) + (e & 1);
}
__device__ __forceinline__ float leakyf(float v)   { return v > 0.f ? v : NEG_SLOPE * v; }
__device__ __forceinline__ float sigmoidf(float v) { return 1.f / (1.f + __expf(-v)); }

#if USE_TDM
// Issue one TDM 2-D tile load: rows x cols f32 elements, row stride
// `stride` elements, from gptr into LDS byte address lds_addr.
// D# layout per CDNA5 ISA 8.3/8.4 (group0 128b, group1 256b).
__device__ __forceinline__ void tdm_load_2d_f32(unsigned lds_addr, const float* gptr,
                                                int rows, int cols,
                                                int tdim0, int stride) {
  unsigned long long ga = (unsigned long long)(uintptr_t)gptr;
  u32x4 g0;
  g0.x = 1u;                                               // count=1, user D#
  g0.y = lds_addr;                                         // lds_addr [63:32]
  g0.z = (unsigned)(ga & 0xffffffffu);                     // global_addr lo
  g0.w = (unsigned)((ga >> 32) & 0x01ffffffu) | (2u << 30);// global_addr hi | type=2
  i32x8 g1;
  g1[0] = (2 << 16);                                       // data_size=4B; mask=0
  g1[1] = (tdim0 & 0xffff) << 16;                          // tensor_dim0[15:0]
  g1[2] = ((tdim0 >> 16) & 0xffff) | ((rows & 0xffff) << 16); // dim0 hi | tensor_dim1 lo
  g1[3] = (cols & 0xffff) << 16;                           // tile_dim0
  g1[4] = rows & 0xffff;                                   // tile_dim1 (tile_dim2=0)
  g1[5] = stride;                                          // tensor_dim0_stride lo32
  g1[6] = 0;                                               // stride hi | dim1_stride lo
  g1[7] = 0;
  i32x4 z4 = {0, 0, 0, 0};
#if __clang_major__ >= 23
  i32x8 z8 = {0, 0, 0, 0, 0, 0, 0, 0};
  __builtin_amdgcn_tensor_load_to_lds(g0, g1, z4, z4, z8, 0);
#else
  __builtin_amdgcn_tensor_load_to_lds(g0, g1, z4, z4, 0);
#endif
}
#endif

// ---------------------------------------------------------------------------
// Implicit-GEMM convolution via v_wmma_f32_16x16x32_bf16.
// One wave computes a 16(OC) x 16(pixel) tile; all 4 waves of a block share
// the same OC tile, so the per-chunk weight tile is staged once in LDS.
// DUAL: accumulate a second conv (same IC/KH/KW) into the same tile.
// ---------------------------------------------------------------------------
template<int IC, int OC, int KH, int KW, int PAD, int ACT, bool DUAL>
__global__ __launch_bounds__(128)
void conv_wmma_kernel(const float* __restrict__ in0, const float* __restrict__ w0,
                      const float* __restrict__ bias0,
                      const float* __restrict__ in1, const float* __restrict__ w1,
                      const float* __restrict__ bias1,
                      float* __restrict__ out) {
  __shared__ __bf16 wtile[DUAL ? 2 : 1][16 * 32];   // per-chunk weight tile(s)

  const int lane = threadIdx.x & 31;
  const int wave = threadIdx.x >> 5;
  const int ntile = blockIdx.x * 4 + wave;          // wave-uniform, grid exact
  const int octile = blockIdx.y * 16;

  // This lane's GEMM column (pixel) for both B fragment and C/D tile.
  const int n   = ntile * 16 + (lane & 15);
  const int bIm = n / HW;
  const int hw  = n % HW;
  const int oy  = hw / Ww;
  const int ox  = hw % Ww;

  constexpr int KTOT = IC * KH * KW;
  v8f acc = {};

  for (int kc = 0; kc < KTOT; kc += 32) {
    // Cooperative stage: 16 rows x 32 k of weights -> LDS (bf16).
    for (int idx = threadIdx.x; idx < 512; idx += 128) {
      const int orow = octile + (idx >> 5);
      const int kk   = kc + (idx & 31);
      float wv = (orow < OC) ? w0[(size_t)orow * KTOT + kk] : 0.f;
      wtile[0][idx] = (__bf16)wv;
      if (DUAL) {
        float wv2 = (orow < OC) ? w1[(size_t)orow * KTOT + kk] : 0.f;
        wtile[1][idx] = (__bf16)wv2;
      }
    }
    if (kc + 64 < KTOT)
      __builtin_prefetch(&w0[(size_t)(octile + (threadIdx.x >> 3)) * KTOT + kc + 64], 0, 1);
    __syncthreads();

    v16bf a, bm;
#pragma unroll
    for (int e = 0; e < 16; ++e) {
      const int koff = koff_of(lane, e);
      const int k = kc + koff;
      a[e] = wtile[0][(lane & 15) * 32 + koff];     // ds_load
      // B: implicit im2col of input
      const int ic = k / (KH * KW);
      const int r  = k % (KH * KW);
      const int iy = oy + (r / KW) - PAD;
      const int ix = ox + (r % KW) - PAD;
      float xv = 0.f;
      if (iy >= 0 && iy < Hh && ix >= 0 && ix < Ww)
        xv = in0[((size_t)(bIm * IC + ic) * Hh + iy) * Ww + ix];
      bm[e] = (__bf16)xv;
    }
    acc = __builtin_amdgcn_wmma_f32_16x16x32_bf16(false, a, false, bm,
                                                  (short)0, acc, false, false);
    if (DUAL) {
      v16bf a2, bm2;
#pragma unroll
      for (int e = 0; e < 16; ++e) {
        const int koff = koff_of(lane, e);
        const int k = kc + koff;
        a2[e] = wtile[1][(lane & 15) * 32 + koff];
        const int ic = k / (KH * KW);
        const int r  = k % (KH * KW);
        const int iy = oy + (r / KW) - PAD;
        const int ix = ox + (r % KW) - PAD;
        float xv = 0.f;
        if (iy >= 0 && iy < Hh && ix >= 0 && ix < Ww)
          xv = in1[((size_t)(bIm * IC + ic) * Hh + iy) * Ww + ix];
        bm2[e] = (__bf16)xv;
      }
      acc = __builtin_amdgcn_wmma_f32_16x16x32_bf16(false, a2, false, bm2,
                                                    (short)0, acc, false, false);
    }
    __syncthreads();   // protect wtile from next-iteration overwrite
  }

  // Epilogue: C/D layout — VGPR i holds M = (lane>=16?8:0)+i, N = lane&15.
#pragma unroll
  for (int i = 0; i < 8; ++i) {
    const int mrow = octile + ((lane & 16) ? 8 : 0) + i;
    if (mrow >= OC) continue;
    float v = acc[i] + bias0[mrow];
    if (DUAL) v += bias1[mrow];
    if (ACT == 1) v = leakyf(v);
    out[((size_t)(bIm * OC + mrow) * Hh + oy) * Ww + ox] = v;
  }
}

// ---------------------------------------------------------------------------
// Precompute bilinear sampling coefficients for warp(prev_h, -flow):
// per (b, link, y, x): 4 clamped plane offsets + 4 weights (0 where invalid).
// ---------------------------------------------------------------------------
__global__ __launch_bounds__(256)
void sample_prep_kernel(const float* __restrict__ flows,   // [B, 2L, H, W]
                        float* __restrict__ sw, int* __restrict__ soff) {
  const int t = blockIdx.x * blockDim.x + threadIdx.x;
  if (t >= Bn * Lc * HW) return;
  const int x = t % Ww;
  const int y = (t / Ww) % Hh;
  const int l = (t / HW) % Lc;
  const int b = t / (Lc * HW);

  const float fx = -flows[((size_t)(b * (2 * Lc) + 2 * l) * Hh + y) * Ww + x];
  const float fy = -flows[((size_t)(b * (2 * Lc) + 2 * l + 1) * Hh + y) * Ww + x];
  // torch grid_sample(align_corners=False) pixel mapping
  const float px = (x + fx) * ((float)Ww / (float)(Ww - 1)) - 0.5f;
  const float py = (y + fy) * ((float)Hh / (float)(Hh - 1)) - 0.5f;
  const float x0f = floorf(px), y0f = floorf(py);
  const float wx = px - x0f, wy = py - y0f;
  const int x0 = (int)x0f, y0 = (int)y0f;

#pragma unroll
  for (int c = 0; c < 4; ++c) {
    const int dx = c & 1, dy = c >> 1;
    const int xi = x0 + dx, yi = y0 + dy;
    const bool valid = (xi >= 0) && (xi < Ww) && (yi >= 0) && (yi < Hh);
    const int xc = min(max(xi, 0), Ww - 1);
    const int yc = min(max(yi, 0), Hh - 1);
    const float wgt = (dx ? wx : 1.f - wx) * (dy ? wy : 1.f - wy);
    sw[4 * (size_t)t + c]  = valid ? wgt : 0.f;
    soff[4 * (size_t)t + c] = yc * Ww + xc;
  }
}

// ---------------------------------------------------------------------------
// Fused h2h 1x1 GEMM (K = L*NF = 832): bilinear gather fused into the
// B-fragment load, 3 gate accumulators (M = c, 64+c, 128+c), full GRU gate
// epilogue. Weight chunks (3 x 16 x 32 f32) are DMA'd into LDS by the
// Tensor Data Mover (wave 0 issues, s_wait_tensorcnt, block barrier).
// Writes next_h to out[0:N] and out[2N:3N].
// ---------------------------------------------------------------------------
__global__ __launch_bounds__(128)
void h2h_gate_kernel(const float* __restrict__ w_ret,   // [192, 832]
                     const float* __restrict__ b_ret,   // [192]
                     const float* __restrict__ prev_h,  // [B, NF, H, W]
                     const float* __restrict__ i2h,     // [B, 192, H, W]
                     const float* __restrict__ sw, const int* __restrict__ soff,
                     float* __restrict__ out) {
  __shared__ float wbuf[3 * 16 * 32];                 // 6 KB: gate-major tiles

  const int lane = threadIdx.x & 31;
  const int wave = threadIdx.x >> 5;
  const int ntile = blockIdx.x * 4 + wave;            // wave-uniform, grid exact
  const int ctile = blockIdx.y * 16;                  // channel tile of NF

  const int n   = ntile * 16 + (lane & 15);
  const int bIm = n / HW;
  const int hw  = n % HW;
  const int oy  = hw / Ww;
  const int ox  = hw % Ww;

  constexpr int KTOT = NFc * Lc;                      // 832 = 26 chunks of 32
  v8f acc0 = {}, acc1 = {}, acc2 = {};

  for (int kc = 0; kc < KTOT; kc += 32) {
#if USE_TDM
    if (wave == 0) {
      // TDM: three 16-row x 32-col f32 tiles (row stride 832) -> LDS.
#pragma unroll
      for (int g = 0; g < 3; ++g) {
        const unsigned lds_addr =
            (unsigned)(uintptr_t)(&wbuf[g * 512]);
        tdm_load_2d_f32(lds_addr,
                        w_ret + (size_t)(g * 64 + ctile) * KTOT + kc,
                        /*rows=*/16, /*cols=*/32,
                        /*tdim0=*/KTOT, /*stride=*/KTOT);
      }
      __builtin_amdgcn_s_wait_tensorcnt(0);
    }
#else
    for (int idx = threadIdx.x; idx < 1536; idx += 128) {
      const int g = idx >> 9;
      const int r = (idx >> 5) & 15;
      const int kk = idx & 31;
      wbuf[idx] = w_ret[(size_t)(g * 64 + ctile + r) * KTOT + kc + kk];
    }
#endif
    __syncthreads();

    // One link per chunk (64 % 32 == 0): load 4 bilinear taps once.
    const int l  = kc >> 6;
    const size_t s4 = 4 * (((size_t)(bIm * Lc + l) * Hh + oy) * Ww + ox);
    const float sw0 = sw[s4], sw1 = sw[s4 + 1], sw2 = sw[s4 + 2], sw3 = sw[s4 + 3];
    const int so0 = soff[s4], so1 = soff[s4 + 1], so2 = soff[s4 + 2], so3 = soff[s4 + 3];

    v16bf bm, a0, a1, a2;
#pragma unroll
    for (int e = 0; e < 16; ++e) {
      const int koff = koff_of(lane, e);
      const int c = (kc & 63) + koff;                 // channel within link
      // B: bilinear sample of prev_h[b, c] (fused warp)
      const float* php = prev_h + (size_t)(bIm * NFc + c) * HW;
      const float v = sw0 * php[so0] + sw1 * php[so1] + sw2 * php[so2] + sw3 * php[so3];
      bm[e] = (__bf16)v;
      // A: staged w_ret rows for the 3 gates (ds_load + cvt)
      const int wi = (lane & 15) * 32 + koff;
      a0[e] = (__bf16)wbuf[wi];
      a1[e] = (__bf16)wbuf[512 + wi];
      a2[e] = (__bf16)wbuf[1024 + wi];
    }
    acc0 = __builtin_amdgcn_wmma_f32_16x16x32_bf16(false, a0, false, bm, (short)0, acc0, false, false);
    acc1 = __builtin_amdgcn_wmma_f32_16x16x32_bf16(false, a1, false, bm, (short)0, acc1, false, false);
    acc2 = __builtin_amdgcn_wmma_f32_16x16x32_bf16(false, a2, false, bm, (short)0, acc2, false, false);
    __syncthreads();   // protect wbuf from next-iteration overwrite
  }

  // GRU gate epilogue, fused.
  const size_t Nh = (size_t)Bn * NFc * HW;
#pragma unroll
  for (int i = 0; i < 8; ++i) {
    const int c = ctile + ((lane & 16) ? 8 : 0) + i;          // 0..63
    const size_t idx = ((size_t)(bIm * NFc + c) * Hh + oy) * Ww + ox;
    const float h0 = acc0[i] + b_ret[c];
    const float h1 = acc1[i] + b_ret[64 + c];
    const float h2 = acc2[i] + b_ret[128 + c];
    const float i0 = i2h[((size_t)(bIm * 192 + c)       * Hh + oy) * Ww + ox];
    const float i1 = i2h[((size_t)(bIm * 192 + 64 + c)  * Hh + oy) * Ww + ox];
    const float i2 = i2h[((size_t)(bIm * 192 + 128 + c) * Hh + oy) * Ww + ox];
    const float reset  = sigmoidf(i0 + h0);
    const float update = sigmoidf(i1 + h1);
    const float newmem = leakyf(i2 + reset * h2);
    const float ph     = prev_h[idx];
    const float nh     = update * ph + (1.f - update) * newmem;
    out[idx]          = nh;   // next_h (output 0)
    out[2 * Nh + idx] = nh;   // next_h (output 2)
  }
}

// ---------------------------------------------------------------------------
extern "C" void kernel_launch(void* const* d_in, const int* in_sizes, int n_in,
                              void* d_out, int out_size, void* d_ws, size_t ws_size,
                              hipStream_t stream) {
  const float* x      = (const float*)d_in[0];
  const float* m      = (const float*)d_in[1];
  const float* prev_h = (const float*)d_in[2];
  const float* w_i2h  = (const float*)d_in[3];
  const float* b_i2h  = (const float*)d_in[4];
  const float* w_i2f  = (const float*)d_in[5];
  const float* b_i2f  = (const float*)d_in[6];
  const float* w_h2f  = (const float*)d_in[7];
  const float* b_h2f  = (const float*)d_in[8];
  const float* w_fl   = (const float*)d_in[9];
  const float* b_fl   = (const float*)d_in[10];
  const float* w_ret  = (const float*)d_in[11];
  const float* b_ret  = (const float*)d_in[12];
  float* out = (float*)d_out;

  // Workspace carve-up (floats): ~104 MB total.
  float* ws_i2h = (float*)d_ws;                       // [B,192,H,W]
  float* ws_f   = ws_i2h + (size_t)Bn * 192 * HW;     // [B, 32,H,W]
  float* ws_fl  = ws_f   + (size_t)Bn * 32  * HW;     // [B, 26,H,W]
  float* ws_sw  = ws_fl  + (size_t)Bn * 26  * HW;     // [B,L,H,W,4] weights
  int*   ws_so  = (int*)(ws_sw + (size_t)Bn * Lc * HW * 4); // offsets

  const int nbx = NTILES / 4;                         // 4 waves per block, exact

  // i2h = conv3x3(x)                         [no act]
  conv_wmma_kernel<64, 192, 3, 3, 1, 0, false>
      <<<dim3(nbx, 12), 128, 0, stream>>>(x, w_i2h, b_i2h, x, w_i2h, b_i2h, ws_i2h);
  // f = leaky(conv5x5(x) + conv5x5(prev_h))  [dual accumulate + leaky]
  conv_wmma_kernel<64, 32, 5, 5, 2, 1, true>
      <<<dim3(nbx, 2), 128, 0, stream>>>(x, w_i2f, b_i2f, prev_h, w_h2f, b_h2f, ws_f);
  // flows = conv5x5(f), 26 channels (2 tiles, store-guarded)
  conv_wmma_kernel<32, 26, 5, 5, 2, 0, false>
      <<<dim3(nbx, 2), 128, 0, stream>>>(ws_f, w_fl, b_fl, ws_f, w_fl, b_fl, ws_fl);
  // Bilinear tap precompute for all 13 links
  sample_prep_kernel<<<(Bn * Lc * HW + 255) / 256, 256, 0, stream>>>(ws_fl, ws_sw, ws_so);
  // Fused warp-gather + TDM-staged h2h GEMM + GRU gates
  h2h_gate_kernel<<<dim3(nbx, 4), 128, 0, stream>>>(w_ret, b_ret, prev_h, ws_i2h,
                                                    ws_sw, ws_so, out);
  // output 1: pass-through m
  hipMemcpyAsync(out + (size_t)Bn * NFc * HW, m,
                 sizeof(float) * (size_t)Bn * NFc * HW,
                 hipMemcpyDeviceToDevice, stream);
}
